// ResNet_64587718197984
// MI455X (gfx1250) — compile-verified, matrix operand order checked by reading, not run on hardware
//
#include <hip/hip_runtime.h>
#include <hip/hip_bf16.h>

// -------------------------------------------------------------------------
// Spiking ResNet-18 forward for MI455X (gfx1250, wave32, WMMA).
// Block convolutions: implicit-GEMM on v_wmma_f32_16x16x32_bf16 (spike
// inputs exact in bf16, f32 accumulate). Workgroup = 8 waves sharing one
// 64-cout N-tile over 8 consecutive 16-pixel M-tiles. Weight chunks
// (4x512 bf16 = 4KB) are double-buffered in LDS via gfx1250 async
// global->LDS DMA (global_load_async_to_lds_b128, gated by
// s_wait_asynccnt): chunk k+1 streams in while 4 WMMAs run on chunk k.
// A fragments are direct contiguous 16-byte global loads from NHWC spikes.
// -------------------------------------------------------------------------

typedef __bf16 bf16_t;
typedef __attribute__((ext_vector_type(16))) __bf16 v16bf;
typedef __attribute__((ext_vector_type(8)))  __bf16 v8bf;
typedef __attribute__((ext_vector_type(8)))  float  v8f;
typedef int v4i __attribute__((vector_size(16)));

#define SNN_THRESH 0.5f
#define SNN_DECAY  0.25f
#define SNN_EPS    1e-5f

#if __has_builtin(__builtin_amdgcn_global_load_async_to_lds_b128) && \
    __has_builtin(__builtin_amdgcn_s_wait_asynccnt)
#define SNN_ASYNC_LDS 1
#else
#define SNN_ASYNC_LDS 0
#endif

// ---------------- stem: direct fp32 conv 7x7 s2 p3, NCHW in -> NHWC out ---
__global__ __launch_bounds__(256)
void snn_stem_conv(const float* __restrict__ x, const float* __restrict__ w,
                   float* __restrict__ out) {
  // x: [32,3,128,128] NCHW, w: [64,3,7,7], out: [32,64,64,64] NHWC
  long idx = (long)blockIdx.x * blockDim.x + threadIdx.x;
  const long total = 32L * 64 * 64 * 64;
  if (idx >= total) return;
  int co = (int)(idx & 63);
  long p  = idx >> 6;
  int ow = (int)(p & 63); p >>= 6;
  int oh = (int)(p & 63); p >>= 6;
  int tb = (int)p;
  float acc = 0.f;
  for (int c = 0; c < 3; ++c) {
    const float* xc = x + ((long)tb * 3 + c) * 128 * 128;
    const float* wc = w + ((long)co * 3 + c) * 49;
    for (int r = 0; r < 7; ++r) {
      int ih = oh * 2 - 3 + r;
      if (ih < 0 || ih >= 128) continue;
      for (int s = 0; s < 7; ++s) {
        int iw = ow * 2 - 3 + s;
        if (iw < 0 || iw >= 128) continue;
        acc += xc[ih * 128 + iw] * wc[r * 7 + s];
      }
    }
  }
  out[idx] = acc;
}

// ---------------- small helpers ------------------------------------------
__global__ void snn_zero(float* __restrict__ p, int n) {
  int i = blockIdx.x * blockDim.x + threadIdx.x;
  if (i < n) p[i] = 0.f;
}

__global__ __launch_bounds__(256)
void snn_bn_stats(const float* __restrict__ x, float* __restrict__ stats,
                  long npix, int C) {
  // NHWC: per-channel sum / sumsq via per-thread strided accumulation.
  int tid = blockIdx.x * blockDim.x + threadIdx.x;   // 131072 threads total
  int c = tid & (C - 1);
  long row = tid / C;
  long nrows = (long)(gridDim.x * blockDim.x) / C;
  float s = 0.f, s2 = 0.f;
  for (; row < npix; row += nrows) {
    float v = x[row * C + c];
    s += v; s2 += v * v;
  }
  atomicAdd(&stats[c], s);
  atomicAdd(&stats[C + c], s2);
}

__global__ void snn_bn_finalize(float* __restrict__ stats,
                                const float* __restrict__ g,
                                const float* __restrict__ b,
                                int C, float invN) {
  int c = threadIdx.x;
  if (c >= C) return;
  float mean = stats[c] * invN;
  float var  = stats[C + c] * invN - mean * mean;
  float sc   = g[c] * rsqrtf(var + SNN_EPS);
  stats[2 * C + c] = sc;
  stats[3 * C + c] = b[c] - mean * sc;
}

__global__ __launch_bounds__(256)
void snn_bn_apply(float* __restrict__ y, const float* __restrict__ stats,
                  long total, int C) {
  long i = (long)blockIdx.x * blockDim.x + threadIdx.x;
  if (i >= total) return;
  int c = (int)(i & (C - 1));
  y[i] = y[i] * stats[2 * C + c] + stats[3 * C + c];
}

// LIF membrane update over T=4 timesteps; emits binary spikes as bf16.
__global__ __launch_bounds__(256)
void snn_mem_update(const float* __restrict__ x, bf16_t* __restrict__ spk,
                    long tstride, int T) {
  long j = (long)blockIdx.x * blockDim.x + threadIdx.x;
  if (j >= tstride) return;
  float mem = 0.f, s = 0.f;
  for (int t = 0; t < T; ++t) {
    float v = x[j + (long)t * tstride];
    mem = mem * SNN_DECAY * (1.f - s) + v;
    s = (mem > SNN_THRESH) ? 1.f : 0.f;
    spk[j + (long)t * tstride] = (bf16_t)s;
  }
}

__global__ __launch_bounds__(256)
void snn_f32_to_bf16(const float* __restrict__ x, bf16_t* __restrict__ y, long n) {
  long i = (long)blockIdx.x * blockDim.x + threadIdx.x;
  if (i < n) y[i] = (bf16_t)x[i];
}

__global__ __launch_bounds__(256)
void snn_add(float* __restrict__ y, const float* __restrict__ a, long n) {
  long i = (long)blockIdx.x * blockDim.x + threadIdx.x;
  if (i < n) y[i] += a[i];
}

// OIHW f32 -> [tap][Cin/32][Cout/16][n=16][k=32] bf16 tiles (B fragments).
__global__ __launch_bounds__(256)
void snn_weight_tile(const float* __restrict__ w, bf16_t* __restrict__ wt,
                     int Cin, int Cout, int RS) {
  long total = (long)RS * Cin * Cout;
  long idx = (long)blockIdx.x * blockDim.x + threadIdx.x;
  if (idx >= total) return;
  int kcN = Cin >> 5, ncN = Cout >> 4;
  int kl  = (int)(idx & 31);
  long r  = idx >> 5;
  int n   = (int)(r & 15); r >>= 4;
  int nc  = (int)(r % ncN); r /= ncN;
  int kc  = (int)(r % kcN); r /= kcN;
  int tap = (int)r;
  int cout = nc * 16 + n;
  int cin  = kc * 32 + kl;
  wt[idx] = (bf16_t)w[((long)cout * Cin + cin) * RS + tap];
}

// ---------------- implicit-GEMM conv via WMMA bf16 -----------------------
// Block = 8 waves x (16 pixels) sharing one 64-cout N-tile. Double-buffered
// LDS weight chunks (async DMA); A fragments loaded direct from global.
__global__ __launch_bounds__(256)
void snn_conv_wmma(const bf16_t* __restrict__ in, const bf16_t* __restrict__ wt,
                   float* __restrict__ out,
                   int TB, int Hin, int Win, int Cin,
                   int Hout, int Wout, int Cout,
                   int stride, int R, int pad) {
  __shared__ bf16_t Bs[2 * 2048];                    // 2 x 4KB B-chunk buffers
  const int tidx = threadIdx.x;
  const int wave = tidx >> 5;
  const int lane = tidx & 31;
  const int tilesN = Cout >> 4;
  const int numNB  = tilesN >> 2;                    // 64-cout groups
  const int kcN = Cin >> 5;
  // block -> (8 consecutive M-tiles, one nb); grids are exact, blocks full.
  int  nb = blockIdx.x % numNB;
  long mTile = (long)(blockIdx.x / numNB) * 8 + wave;
  int cout0 = nb << 6;

  // This lane supplies row m = lane&15 of A, K-chunk selected by lane>>4.
  int m    = lane & 15;
  int koff = (lane >> 4) << 3;                       // 0 or 8 channels
  long M = mTile * 16 + m;
  int ow = (int)(M % Wout);
  long t = M / Wout;
  int oh = (int)(t % Hout);
  int tb = (int)(t / Hout);

  v8f acc[4] = {};
  const int taps = R * R;
  const int totalChunks = taps * kcN;
  const long bChunkStride = (long)tilesN * 512;      // halfs per (tap,kc) step
  const bf16_t* bsrc0 = wt + ((long)(nb << 2)) * 512;

  // stage weight chunk `chnk` into LDS buffer `buf` (16B per thread).
  auto stage = [&](int chnk, int buf) {
    const bf16_t* s = bsrc0 + (long)chnk * bChunkStride + tidx * 8;
    bf16_t* d = Bs + buf * 2048 + tidx * 8;
#if SNN_ASYNC_LDS
    __builtin_amdgcn_global_load_async_to_lds_b128(
        (__attribute__((address_space(1))) v4i*)(void*)(s),
        (__attribute__((address_space(3))) v4i*)(void*)(d), 0, 0);
#else
    *(v8bf*)d = *(const v8bf*)s;
#endif
  };

  stage(0, 0);
  int chunk = 0;
  for (int tap = 0; tap < taps; ++tap) {
    int r = tap / R, s = tap - r * R;
    int ih = oh * stride - pad + r;
    int iw = ow * stride - pad + s;
    bool inb = (ih >= 0) && (ih < Hin) && (iw >= 0) && (iw < Win);
    const bf16_t* ap = in + (((long)tb * Hin + ih) * Win + iw) * Cin + koff;
    for (int kc = 0; kc < kcN; ++kc, ++chunk) {
      int cur = chunk & 1;
      // kick off next chunk into the other buffer while we compute
      if (chunk + 1 < totalChunks) {
        stage(chunk + 1, cur ^ 1);
#if SNN_ASYNC_LDS
        __builtin_amdgcn_s_wait_asynccnt(1);         // chunk k landed
#endif
      } else {
#if SNN_ASYNC_LDS
        __builtin_amdgcn_s_wait_asynccnt(0);
#endif
      }
      __syncthreads();                               // chunk k visible to WG
      // ---- A fragment: direct global load (zero rows for padding) ----
      v16bf a = {};
      if (inb) {
        const bf16_t* p = ap + kc * 32;
        union { v16bf v; v8bf h[2]; } ua;
        ua.h[0] = *(const v8bf*)(p);                 // K 0..7   (or 8..15)
        ua.h[1] = *(const v8bf*)(p + 16);            // K 16..23 (or 24..31)
        a = ua.v;
      }
      // ---- 4 back-to-back WMMAs, B fragments from LDS ----
      const bf16_t* bbase = Bs + cur * 2048 + (lane & 15) * 32 + koff;
#pragma unroll
      for (int j = 0; j < 4; ++j) {
        const bf16_t* q = bbase + j * 512;
        union { v16bf v; v8bf h[2]; } ub;
        ub.h[0] = *(const v8bf*)(q);
        ub.h[1] = *(const v8bf*)(q + 16);
        acc[j] = __builtin_amdgcn_wmma_f32_16x16x32_bf16(
            false, a, false, ub.v, (short)0, acc[j], false, false);
      }
      __syncthreads();                               // done reading buf `cur`
    }
  }

  // D layout: vgpr i -> row m = i + 8*(lane>=16); col n = lane&15.
  int n = lane & 15;
  int rbase = (lane >> 4) << 3;
  long rowOfs[8];
#pragma unroll
  for (int i = 0; i < 8; ++i) {
    long Mi = mTile * 16 + rbase + i;
    int ow2 = (int)(Mi % Wout);
    long t2 = Mi / Wout;
    int oh2 = (int)(t2 % Hout);
    int tb2 = (int)(t2 / Hout);
    rowOfs[i] = (((long)tb2 * Hout + oh2) * Wout + ow2) * Cout;
  }
#pragma unroll
  for (int j = 0; j < 4; ++j) {
#pragma unroll
    for (int i = 0; i < 8; ++i) {
      out[rowOfs[i] + cout0 + (j << 4) + n] = acc[j][i];
    }
  }
}

// ---------------- pooling + FC -------------------------------------------
__global__ void snn_pool(const float* __restrict__ x, float* __restrict__ out) {
  // x: [32,8,8,512] NHWC, out: [8,512]; mean over T,H,W (=256 elems).
  int tid = blockIdx.x * blockDim.x + threadIdx.x;
  if (tid >= 8 * 512) return;
  int c = tid & 511;
  int b = tid >> 9;
  float s = 0.f;
  for (int t = 0; t < 4; ++t)
    for (int p = 0; p < 64; ++p)
      s += x[(((long)(t * 8 + b)) * 64 + p) * 512 + c];
  out[tid] = s * (1.f / 256.f);
}

__global__ void snn_fc(const float* __restrict__ pool, const float* __restrict__ w,
                       const float* __restrict__ bias, float* __restrict__ out) {
  int tid = blockIdx.x * blockDim.x + threadIdx.x;
  if (tid >= 8 * 100) return;
  int j = tid % 100, b = tid / 100;
  float s = bias[j];
  for (int k = 0; k < 512; ++k) s += pool[b * 512 + k] * w[j * 512 + k];
  out[b * 100 + j] = s;
}

// -------------------------------------------------------------------------
extern "C" void kernel_launch(void* const* d_in, const int* in_sizes, int n_in,
                              void* d_out, int out_size, void* d_ws, size_t ws_size,
                              hipStream_t stream) {
  (void)in_sizes; (void)n_in; (void)out_size; (void)ws_size;
  const int cfg[8][3] = {{64,64,1},{64,64,1},{64,128,2},{128,128,1},
                         {128,256,2},{256,256,1},{256,512,2},{512,512,1}};
  int ii = 0;
  const float* x      = (const float*)d_in[ii++];
  const float* stem_w = (const float*)d_in[ii++];
  const float* stem_g = (const float*)d_in[ii++];
  const float* stem_b = (const float*)d_in[ii++];
  struct Blk { const float *w1,*g1,*b1,*w2,*g2,*b2,*ws,*gs,*bs; bool sc; };
  Blk blk[8];
  for (int i = 0; i < 8; ++i) {
    blk[i].w1 = (const float*)d_in[ii++];
    blk[i].g1 = (const float*)d_in[ii++];
    blk[i].b1 = (const float*)d_in[ii++];
    blk[i].w2 = (const float*)d_in[ii++];
    blk[i].g2 = (const float*)d_in[ii++];
    blk[i].b2 = (const float*)d_in[ii++];
    blk[i].sc = (cfg[i][2] != 1) || (cfg[i][0] != cfg[i][1]);
    if (blk[i].sc) {
      blk[i].ws = (const float*)d_in[ii++];
      blk[i].gs = (const float*)d_in[ii++];
      blk[i].bs = (const float*)d_in[ii++];
    } else { blk[i].ws = blk[i].gs = blk[i].bs = nullptr; }
  }
  const float* fc_w = (const float*)d_in[ii++];
  const float* fc_b = (const float*)d_in[ii++];

  // ---- carve workspace (256B aligned regions) ----
  char* wsp = (char*)d_ws;
  auto carve = [&](size_t bytes) -> void* {
    void* r = (void*)wsp; wsp += (bytes + 255) & ~(size_t)255; return r;
  };
  const long MAXF = 32L * 64 * 64 * 64;              // 8,388,608 elems
  float*  B0   = (float*)carve(MAXF * 4);
  float*  B1   = (float*)carve(MAXF * 4);
  float*  B2   = (float*)carve(MAXF * 4);
  float*  SCb  = (float*)carve((size_t)(32L * 32 * 32 * 128) * 4);
  bf16_t* SPK  = (bf16_t*)carve(MAXF * 2);
  bf16_t* XB   = (bf16_t*)carve(MAXF * 2);
  bf16_t* WT   = (bf16_t*)carve((size_t)(9L * 512 * 512) * 2);
  float*  STAT = (float*)carve(4 * 512 * 4);
  float*  POOL = (float*)carve(8 * 512 * 4);

  auto run_bn = [&](float* buf, long npix, int C, const float* g, const float* b) {
    snn_zero<<<1, 1024, 0, stream>>>(STAT, 2 * C);
    snn_bn_stats<<<512, 256, 0, stream>>>(buf, STAT, npix, C);
    snn_bn_finalize<<<1, 512, 0, stream>>>(STAT, g, b, C, 1.f / (float)npix);
    long total = npix * C;
    snn_bn_apply<<<(unsigned)((total + 255) / 256), 256, 0, stream>>>(buf, STAT, total, C);
  };

  auto run_conv = [&](const bf16_t* in, const float* w, float* out,
                      int Hin, int Cin, int Hout, int Cout,
                      int stride, int R, int pad) {
    int RS = R * R;
    long wtot = (long)RS * Cin * Cout;
    snn_weight_tile<<<(unsigned)((wtot + 255) / 256), 256, 0, stream>>>(w, WT, Cin, Cout, RS);
    long numMtiles = (long)32 * Hout * Hout >> 4;    // multiple of 8
    long blocks = (numMtiles >> 3) * (Cout >> 6);    // 8 M-tiles x 64 couts per block
    snn_conv_wmma<<<(unsigned)blocks, 256, 0, stream>>>(
        in, WT, out, 32, Hin, Hin, Cin, Hout, Hout, Cout, stride, R, pad);
  };

  // ---- stem: conv7x7 s2 p3 + TDBN ----
  {
    long total = 32L * 64 * 64 * 64;
    snn_stem_conv<<<(unsigned)((total + 255) / 256), 256, 0, stream>>>(x, stem_w, B0);
    run_bn(B0, 32L * 64 * 64, 64, stem_g, stem_b);
  }

  // ---- 8 BasicBlocks ----
  int H = 64;
  float *X = B0, *T1 = B1, *T2 = B2;
  for (int i = 0; i < 8; ++i) {
    int Cin = cfg[i][0], Cout = cfg[i][1], s = cfg[i][2];
    int Ho = H / s;
    long inTot  = 32L * H  * H  * Cin;
    long outTot = 32L * Ho * Ho * Cout;

    long ts1 = inTot / 4;   // per-timestep elements (B*H*W*C)
    snn_mem_update<<<(unsigned)((ts1 + 255) / 256), 256, 0, stream>>>(X, SPK, ts1, 4);
    run_conv(SPK, blk[i].w1, T1, H, Cin, Ho, Cout, s, 3, 1);
    run_bn(T1, 32L * Ho * Ho, Cout, blk[i].g1, blk[i].b1);

    long ts2 = outTot / 4;
    snn_mem_update<<<(unsigned)((ts2 + 255) / 256), 256, 0, stream>>>(T1, SPK, ts2, 4);
    run_conv(SPK, blk[i].w2, T2, Ho, Cout, Ho, Cout, 1, 3, 1);
    run_bn(T2, 32L * Ho * Ho, Cout, blk[i].g2, blk[i].b2);

    if (blk[i].sc) {
      snn_f32_to_bf16<<<(unsigned)((inTot + 255) / 256), 256, 0, stream>>>(X, XB, inTot);
      run_conv(XB, blk[i].ws, SCb, H, Cin, Ho, Cout, s, 1, 0);
      run_bn(SCb, 32L * Ho * Ho, Cout, blk[i].gs, blk[i].bs);
      snn_add<<<(unsigned)((outTot + 255) / 256), 256, 0, stream>>>(T2, SCb, outTot);
    } else {
      snn_add<<<(unsigned)((outTot + 255) / 256), 256, 0, stream>>>(T2, X, outTot);
    }
    float* oldX = X; X = T2; T2 = oldX;  // rotate; T1 stays scratch
    H = Ho;
  }

  // ---- pool + FC ----
  snn_pool<<<16, 256, 0, stream>>>(X, POOL);
  snn_fc<<<4, 256, 0, stream>>>(POOL, fc_w, fc_b, (float*)d_out);
}
